// MuSc_5703716569671
// MI455X (gfx1250) — matrix operand order repaired and checked by reading, not compiled, not run on hardware
//
#include <hip/hip_runtime.h>
#include <hip/hip_bf16.h>
#include <math.h>

// ---------------- problem constants ----------------
#define LL    4          // layers
#define BI    8          // batch
#define PH    37
#define PW    37
#define PP    1369       // patches
#define DD    1024       // feature dim
#define JJ    768        // cls dim
#define HH    518
#define WW    518
#define PPAD  1408       // padded patch count (11 * 128)
#define NPAIR 28         // unordered pairs i<b
#define QG    22         // q groups of 64 (22*64 = 1408)
#define QCAP  1376       // per-(l,pair) column-max slots (multiple of 16, >= PP)

typedef _Float16 v8h  __attribute__((ext_vector_type(8)));
typedef _Float16 v16h __attribute__((ext_vector_type(16)));
typedef float    v8f  __attribute__((ext_vector_type(8)));

// ---------------- workspace layout (bytes) ----------------
#define LN_BYTES    ((size_t)LL * BI * PP * DD * 4)          // 179,437,568
#define X_BYTES     ((size_t)LL * BI * PPAD * DD * 2)        //  92,274,688
#define MIND_BYTES  ((size_t)LL * BI * PP * BI * 4)          //   1,401,856
#define SCORE_BYTES ((size_t)44032)                          // BI*PP*4 rounded up
#define IMG_BYTES   ((size_t)256)
#define QBUF_ELEMS  ((size_t)LL * NPAIR * QCAP)              // 154,112
// dmaxQ (uint) after img

// order-preserving float<->uint encoding for atomicMax
__device__ __forceinline__ unsigned int enc_f(float f) {
  unsigned int u = __float_as_uint(f);
  return (u & 0x80000000u) ? ~u : (u | 0x80000000u);
}
__device__ __forceinline__ float dec_f(unsigned int u) {
  return (u & 0x80000000u) ? __uint_as_float(u ^ 0x80000000u) : __uint_as_float(~u);
}
__device__ __forceinline__ void pair_ib(int pair, int& i, int& b) {
  int i0 = 0, c = BI - 1;
  while (pair >= c) { pair -= c; ++i0; --c; }
  i = i0; b = i0 + 1 + pair;
}

// ---------------- 0: zero f16 pad rows + zero score accumulator ----------------
__global__ __launch_bounds__(256) void musc_init(_Float16* __restrict__ X,
                                                 float* __restrict__ scores) {
  long long idx = (long long)blockIdx.x * 256 + threadIdx.x;
  const long long padPerLB = (long long)(PPAD - PP) * DD;
  const long long padElems = (long long)LL * BI * padPerLB;
  if (idx < padElems) {
    long long lb  = idx / padPerLB;
    long long rem = idx % padPerLB;
    X[lb * (long long)PPAD * DD + (long long)PP * DD + rem] = (_Float16)0.0f;
  }
  if (idx < (long long)BI * PP) scores[idx] = 0.0f;
}

// zero the column-max buffer (per r iteration)
__global__ __launch_bounds__(256) void musc_initq(unsigned int* __restrict__ q) {
  const long long idx = (long long)blockIdx.x * 256 + threadIdx.x;
  if (idx < (long long)QBUF_ELEMS) q[idx] = 0u;
}

// ---------------- 1: LayerNorm over (P,D) jointly per (l,b) ----------------
__global__ __launch_bounds__(256) void musc_ln(const float* __restrict__ in,
                                               float* __restrict__ out) {
  const int lb = blockIdx.x;                 // 0..31
  const size_t N = (size_t)PP * DD;
  const float* src = in + (size_t)lb * N;
  float* dst = out + (size_t)lb * N;
  const int t = threadIdx.x;
  double s = 0.0, s2 = 0.0;
  for (size_t i = t; i < N; i += 256) { float v = src[i]; s += v; s2 += (double)v * v; }
  __shared__ double rs[256], rs2[256];
  rs[t] = s; rs2[t] = s2; __syncthreads();
  for (int off = 128; off > 0; off >>= 1) {
    if (t < off) { rs[t] += rs[t + off]; rs2[t] += rs2[t + off]; }
    __syncthreads();
  }
  __shared__ float mu_s, rstd_s;
  if (t == 0) {
    double mu  = rs[0]  / (double)N;
    double var = rs2[0] / (double)N - mu * mu;
    mu_s = (float)mu;
    rstd_s = (float)(1.0 / sqrt(var + 1e-5));
  }
  __syncthreads();
  const float mu = mu_s, rstd = rstd_s;
  for (size_t i = t; i < N; i += 256) dst[i] = (src[i] - mu) * rstd;
}

// ---------------- 2: r x r box average + L2 normalize + f16 convert ----------------
__global__ __launch_bounds__(256) void musc_boxnorm(const float* __restrict__ ln,
                                                    _Float16* __restrict__ X, int r) {
  const int gid = blockIdx.x;            // l*B*P + b*P + p
  const int p  = gid % PP;
  const int lb = gid / PP;
  const int ph = p / PW, pw = p % PW;
  const float* src = ln + (size_t)lb * PP * DD;
  _Float16* dst = X + (size_t)lb * PPAD * DD + (size_t)p * DD;
  const int t = threadIdx.x;
  const int pad = r / 2;
  const float inv_rr = 1.0f / (float)(r * r);
  float vals[4];
  float ss = 0.0f;
#pragma unroll
  for (int s = 0; s < 4; ++s) {
    const int d = t + 256 * s;
    float acc = 0.0f;
    for (int dy = -pad; dy <= pad; ++dy) {
      int y = ph + dy; if (y < 0 || y >= PH) continue;
      for (int dx = -pad; dx <= pad; ++dx) {
        int x = pw + dx; if (x < 0 || x >= PW) continue;
        acc += src[(size_t)(y * PW + x) * DD + d];
      }
    }
    acc *= inv_rr;
    vals[s] = acc;
    ss += acc * acc;
  }
  __shared__ float red[256];
  red[t] = ss; __syncthreads();
  for (int off = 128; off > 0; off >>= 1) {
    if (t < off) red[t] += red[t + off];
    __syncthreads();
  }
  __shared__ float inv_s;
  if (t == 0) inv_s = 1.0f / sqrtf(red[0]);
  __syncthreads();
  const float inv = inv_s;
#pragma unroll
  for (int s = 0; s < 4; ++s) dst[t + 256 * s] = (_Float16)(vals[s] * inv);
}

// ---------------- 3: WMMA GEMM, both reduction directions ----------------
// grid: (11 p-blocks, 28 unordered pairs i<b, 4 layers); block: 128 = 4 waves.
// Each wave: 2 M-tiles (32 p rows) x 4 N-tiles (64 q cols) register tiling,
// sweeping 22 q-groups x 32 K-steps.  Row direction (min over q for (i,p,b))
// reduced in-wave exactly; column direction (min over p for (b,q,i)) combined
// across p-blocks via atomicMax on an order-preserving u32 encoding.
__global__ __launch_bounds__(128) void musc_gemm_min(const _Float16* __restrict__ X,
                                                     float* __restrict__ mind,
                                                     unsigned int* __restrict__ dq) {
  const int l    = blockIdx.z;
  const int pair = blockIdx.y;               // 0..27
  int i, b; pair_ib(pair, i, b);             // i < b
  const int wave = threadIdx.x >> 5;         // 0..3
  const int lane = threadIdx.x & 31;
  const int pbase = blockIdx.x * 128 + wave * 32;   // two 16-row tiles

  const _Float16* A  = X + ((size_t)l * BI + i) * PPAD * DD;
  const _Float16* Bm = X + ((size_t)l * BI + b) * PPAD * DD;

  // A fragment addressing (16x32 f16): lane&15 = row m, lane>>4 selects K phase
  const int mrow  = lane & 15;
  const int kselA = (lane >> 4) * 8;          // halves: v0..3 at +0, v4..7 at +16
  const _Float16* aptr0 = A + (size_t)(pbase + mrow) * DD + kselA;
  const _Float16* aptr1 = aptr0 + (size_t)16 * DD;
  // B fragment addressing (32x16 f16): lane&15 = col n (=q), halves at +0 and +16 (same K phase as A)
  const int nlane = lane & 15;
  const int kselB = (lane >> 4) * 8;
  const int mofs  = (lane >> 4) * 8;          // row offset held by this half-group

  float dmax0[8], dmax1[8];
#pragma unroll
  for (int j = 0; j < 8; ++j) { dmax0[j] = -3.0e38f; dmax1[j] = -3.0e38f; }

  unsigned int* dqbase = dq + ((size_t)l * NPAIR + pair) * QCAP;

  for (int qg = 0; qg < QG; ++qg) {
    const int qbase = qg * 64;
    v8f c0[4], c1[4];
#pragma unroll
    for (int s = 0; s < 4; ++s) { c0[s] = (v8f){}; c1[s] = (v8f){}; }

    for (int kk = 0; kk < DD; kk += 32) {
      union { v16h v; v8h h[2]; } ua0, ua1;
      ua0.h[0] = *(const v8h*)(aptr0 + kk);
      ua0.h[1] = *(const v8h*)(aptr0 + kk + 16);
      ua1.h[0] = *(const v8h*)(aptr1 + kk);
      ua1.h[1] = *(const v8h*)(aptr1 + kk + 16);
#pragma unroll
      for (int s = 0; s < 4; ++s) {
        const _Float16* bp = Bm + (size_t)(qbase + s * 16 + nlane) * DD + kselB + kk;
        union { v16h v; v8h h[2]; } ub;
        ub.h[0] = *(const v8h*)(bp);
        ub.h[1] = *(const v8h*)(bp + 16);
        c0[s] = __builtin_amdgcn_wmma_f32_16x16x32_f16(false, ua0.v, false, ub.v,
                                                       (short)0, c0[s], false, false);
        c1[s] = __builtin_amdgcn_wmma_f32_16x16x32_f16(false, ua1.v, false, ub.v,
                                                       (short)0, c1[s], false, false);
      }
    }

#pragma unroll
    for (int s = 0; s < 4; ++s) {
      const int qv = qbase + s * 16 + nlane;
      const bool qok = (qv < PP);
      // row direction: running max(dot) per (m-row, this lane's n)
      if (qok) {
#pragma unroll
        for (int j = 0; j < 8; ++j) {
          dmax0[j] = fmaxf(dmax0[j], c0[s][j]);
          dmax1[j] = fmaxf(dmax1[j], c1[s][j]);
        }
      }
      // column direction: max over valid p rows of this 32-row strip
      float cm = -3.0e38f;
#pragma unroll
      for (int j = 0; j < 8; ++j) {
        if (pbase + mofs + j < PP)      cm = fmaxf(cm, c0[s][j]);
        if (pbase + 16 + mofs + j < PP) cm = fmaxf(cm, c1[s][j]);
      }
      cm = fmaxf(cm, __shfl_xor(cm, 16, 32));   // merge other half's m rows
      if (qok && lane < 16)
        atomicMax(dqbase + qv, enc_f(cm));
    }
  }

  // row reduction: max over n within each 16-lane half-group (wave32)
#pragma unroll
  for (int j = 0; j < 8; ++j) {
    float v0 = dmax0[j], v1 = dmax1[j];
    v0 = fmaxf(v0, __shfl_xor(v0, 1, 32));
    v0 = fmaxf(v0, __shfl_xor(v0, 2, 32));
    v0 = fmaxf(v0, __shfl_xor(v0, 4, 32));
    v0 = fmaxf(v0, __shfl_xor(v0, 8, 32));
    v1 = fmaxf(v1, __shfl_xor(v1, 1, 32));
    v1 = fmaxf(v1, __shfl_xor(v1, 2, 32));
    v1 = fmaxf(v1, __shfl_xor(v1, 4, 32));
    v1 = fmaxf(v1, __shfl_xor(v1, 8, 32));
    dmax0[j] = v0; dmax1[j] = v1;
  }
  if ((lane & 15) == 0) {
    const int mo = (lane >> 4) * 8;             // lane0 -> m=0..7, lane16 -> m=8..15
#pragma unroll
    for (int j = 0; j < 8; ++j) {
      const int p0 = pbase + mo + j;
      if (p0 < PP) {
        float d2 = fmaxf(2.0f - 2.0f * dmax0[j], 1e-12f);
        mind[(((size_t)l * BI + i) * PP + p0) * BI + b] = sqrtf(d2);
      }
      const int p1 = pbase + 16 + mo + j;
      if (p1 < PP) {
        float d2 = fmaxf(2.0f - 2.0f * dmax1[j], 1e-12f);
        mind[(((size_t)l * BI + i) * PP + p1) * BI + b] = sqrtf(d2);
      }
    }
  }
}

// ---------------- 3b: decode column-direction maxima -> mind[l][b][q][i] ----------------
__global__ __launch_bounds__(256) void musc_colmind(const unsigned int* __restrict__ dq,
                                                    float* __restrict__ mind) {
  const int idx = blockIdx.x * 256 + threadIdx.x;      // over LL*NPAIR*PP
  if (idx >= LL * NPAIR * PP) return;
  const int q    = idx % PP;
  const int rest = idx / PP;
  const int pair = rest % NPAIR;
  const int l    = rest / NPAIR;
  int i, b; pair_ib(pair, i, b);
  const float f = dec_f(dq[((size_t)l * NPAIR + pair) * QCAP + q]);
  const float d2 = fmaxf(2.0f - 2.0f * f, 1e-12f);
  mind[(((size_t)l * BI + b) * PP + q) * BI + i] = sqrtf(d2);
}

// ---------------- 4: mean of 2 smallest cross-image minima, accumulate ----------------
__global__ __launch_bounds__(256) void musc_score(const float* __restrict__ mind,
                                                  float* __restrict__ scores,
                                                  float weight) {
  const int idx = blockIdx.x * 256 + threadIdx.x;   // over B*P
  if (idx >= BI * PP) return;
  const int i = idx / PP, p = idx % PP;
  float acc = 0.0f;
  for (int l = 0; l < LL; ++l) {
    const float* m = mind + (((size_t)l * BI + i) * PP + p) * BI;
    float m1 = 3.0e38f, m2 = 3.0e38f;
#pragma unroll
    for (int b = 0; b < BI; ++b) {
      if (b == i) continue;
      const float v = m[b];
      if (v < m1) { m2 = m1; m1 = v; }
      else if (v < m2) { m2 = v; }
    }
    acc += 0.5f * (m1 + m2);
  }
  scores[idx] += weight * acc;
}

// ---------------- 5: per-image max ----------------
__global__ __launch_bounds__(256) void musc_imgmax(const float* __restrict__ scores,
                                                   float* __restrict__ img) {
  const int i = blockIdx.x, t = threadIdx.x;
  float m = -3.0e38f;
  for (int p = t; p < PP; p += 256) m = fmaxf(m, scores[i * PP + p]);
  __shared__ float red[256];
  red[t] = m; __syncthreads();
  for (int off = 128; off > 0; off >>= 1) {
    if (t < off) red[t] = fmaxf(red[t], red[t + off]);
    __syncthreads();
  }
  if (t == 0) img[i] = red[0];
}

// ---------------- 6: cls similarity + top-k re-scoring -> final(8) ----------------
__global__ __launch_bounds__(64) void musc_final(const float* __restrict__ cls,
                                                 const float* __restrict__ img,
                                                 float* __restrict__ out) {
  __shared__ float nrm[8];
  __shared__ float sim[8][8];
  __shared__ float s_img[8];
  const int t = threadIdx.x;
  if (t < 8) {
    const float* row = cls + t * JJ;
    float s = 0.0f;
    for (int j = 0; j < JJ; ++j) { float v = row[j]; s += v * v; }
    nrm[t] = sqrtf(s);
    s_img[t] = img[t];
  }
  __syncthreads();
  if (t < 64) {
    const int i = t >> 3, j = t & 7;
    const float* ri = cls + i * JJ;
    const float* rj = cls + j * JJ;
    float d = 0.0f;
    for (int k = 0; k < JJ; ++k) d += ri[k] * rj[k];
    sim[i][j] = d / (nrm[i] * nrm[j]);
  }
  __syncthreads();
  if (t < 8) {
    float a[8]; bool used[8];
    for (int j = 0; j < 8; ++j) { a[j] = sim[t][j]; used[j] = false; }
    float selv[3]; int seli[3];
    for (int m = 0; m < 3; ++m) {
      int bi = 0; float bv = -3.0e38f;
      for (int j = 0; j < 8; ++j)
        if (!used[j] && a[j] > bv) { bv = a[j]; bi = j; }
      used[bi] = true; selv[m] = bv; seli[m] = bi;
    }
    float res = 0.0f, sum = 0.0f, dot = 0.0f;
    for (int k = 0; k < 3; ++k) {          // k = top-1,2,3
      sum += selv[k];
      dot += selv[k] * s_img[seli[k]];
      res += dot / sum;
    }
    out[t] = res / 3.0f;
  }
}

// ---------------- 7: bilinear upsample 37x37 -> 518x518 ----------------
__global__ __launch_bounds__(256) void musc_pixel(const float* __restrict__ scores,
                                                  float* __restrict__ out) {
  const long long total = (long long)BI * HH * WW;
  long long idx = (long long)blockIdx.x * 256 + threadIdx.x;
  if (idx >= total) return;
  const int x = (int)(idx % WW);
  const long long r1 = idx / WW;
  const int y = (int)(r1 % HH);
  const int b = (int)(r1 / HH);
  const float sc = 36.0f / 517.0f;
  const float py = (float)y * sc, px = (float)x * sc;
  int iy0 = (int)floorf(py); iy0 = min(max(iy0, 0), PH - 1);
  int ix0 = (int)floorf(px); ix0 = min(max(ix0, 0), PW - 1);
  const int iy1 = min(iy0 + 1, PH - 1);
  const int ix1 = min(ix0 + 1, PW - 1);
  const float wy = py - (float)iy0, wx = px - (float)ix0;
  const float* S = scores + (size_t)b * PP;
  const float v00 = S[iy0 * PW + ix0], v01 = S[iy0 * PW + ix1];
  const float v10 = S[iy1 * PW + ix0], v11 = S[iy1 * PW + ix1];
  const float v = (1.0f - wy) * ((1.0f - wx) * v00 + wx * v01)
                +          wy * ((1.0f - wx) * v10 + wx * v11);
  out[8 + idx] = v;
}

// ---------------- launcher ----------------
extern "C" void kernel_launch(void* const* d_in, const int* in_sizes, int n_in,
                              void* d_out, int out_size, void* d_ws, size_t ws_size,
                              hipStream_t stream) {
  const float* features = (const float*)d_in[0];   // (4,8,1369,1024) f32
  const float* cls      = (const float*)d_in[1];   // (8,768) f32
  float* out = (float*)d_out;                      // final(8) ++ pixel(8*518*518)

  char* ws = (char*)d_ws;
  float*        ln     = (float*)ws;
  _Float16*     X      = (_Float16*)(ws + LN_BYTES);
  float*        mind   = (float*)(ws + LN_BYTES + X_BYTES);
  float*        scores = (float*)(ws + LN_BYTES + X_BYTES + MIND_BYTES);
  float*        img    = (float*)(ws + LN_BYTES + X_BYTES + MIND_BYTES + SCORE_BYTES);
  unsigned int* dmaxQ  = (unsigned int*)(ws + LN_BYTES + X_BYTES + MIND_BYTES +
                                         SCORE_BYTES + IMG_BYTES);

  // 0: zero f16 pad rows + score accumulator
  {
    const long long padElems = (long long)LL * BI * (PPAD - PP) * DD; // 1,277,952
    musc_init<<<(int)((padElems + 255) / 256), 256, 0, stream>>>(X, scores);
  }
  // 1: layer norm
  musc_ln<<<LL * BI, 256, 0, stream>>>(features, ln);

  // 2-4: per receptive-field radius
  const int r_list[3] = {1, 3, 5};
  for (int ri = 0; ri < 3; ++ri) {
    musc_boxnorm<<<LL * BI * PP, 256, 0, stream>>>(ln, X, r_list[ri]);
    musc_initq<<<(int)((QBUF_ELEMS + 255) / 256), 256, 0, stream>>>(dmaxQ);
    dim3 grid(PPAD / 128, NPAIR, LL);            // (11, 28, 4)
    musc_gemm_min<<<grid, 128, 0, stream>>>(X, mind, dmaxQ);
    musc_colmind<<<(LL * NPAIR * PP + 255) / 256, 256, 0, stream>>>(dmaxQ, mind);
    musc_score<<<(BI * PP + 255) / 256, 256, 0, stream>>>(mind, scores,
                                                          1.0f / (float)(LL * 3));
  }

  // 5: image-level scores
  musc_imgmax<<<BI, 256, 0, stream>>>(scores, img);
  // 6: cls re-scoring -> out[0..7]
  musc_final<<<1, 64, 0, stream>>>(cls, img, out);
  // 7: bilinear upsample -> out[8..]
  {
    const long long total = (long long)BI * HH * WW;
    musc_pixel<<<(int)((total + 255) / 256), 256, 0, stream>>>(scores, out);
  }
}
